// QuantESPCNV2_69612829934278
// MI455X (gfx1250) — 8-bit matrix operand order repaired and checked by reading, compiled, not run on hardware
//
#include <hip/hip_runtime.h>
#include <stdint.h>

typedef __attribute__((ext_vector_type(8))) int v8i;

#define S_IO   0.02f
#define S_ACT  0.1f
#define BN_EPS 1e-5f

static constexpr size_t OFF_ACT1 = 0;
static constexpr size_t SZ_ACT1  = (size_t)4 * 258 * 258 * 64;
static constexpr size_t OFF_ACT2 = OFF_ACT1 + SZ_ACT1;
static constexpr size_t SZ_ACT2  = SZ_ACT1;
static constexpr size_t OFF_ACT3 = OFF_ACT2 + SZ_ACT2;
static constexpr size_t SZ_ACT3  = (size_t)4 * 258 * 258 * 32;
static constexpr size_t OFF_W1D  = OFF_ACT3 + SZ_ACT3;
static constexpr size_t OFF_W2P  = OFF_W1D + 19456;
static constexpr size_t OFF_S2   = OFF_W2P + 36864;
static constexpr size_t OFF_W3P  = OFF_S2 + 256;
static constexpr size_t OFF_S3   = OFF_W3P + 18432;
static constexpr size_t OFF_W4Q  = OFF_S3 + 256;
static constexpr size_t OFF_S4   = OFF_W4Q + 1024;
static constexpr size_t ZERO_BYTES = OFF_ACT3 + SZ_ACT3;

__global__ void k_zero(uint4* __restrict__ p, size_t n16) {
    size_t i = (size_t)blockIdx.x * blockDim.x + threadIdx.x;
    if (i < n16) p[i] = make_uint4(0u, 0u, 0u, 0u);
}

__global__ void k_w1_dequant(const float* __restrict__ w, float* __restrict__ wd) {
    int oc = threadIdx.x;
    const float* src = w + oc * 75;
    float mx = 0.f;
    for (int i = 0; i < 75; i++) mx = fmaxf(mx, fabsf(src[i]));
    float s = mx / 7.0f + 1e-12f;
    for (int i = 0; i < 75; i++) {
        float q = fminf(fmaxf(rintf(src[i] / s), -7.f), 7.f);
        wd[oc * 75 + i] = q * s;
    }
}

__global__ void k_wscale(const float* __restrict__ w, float* __restrict__ sc, int K) {
    int oc = threadIdx.x;
    const float* src = w + (size_t)oc * K;
    float mx = 0.f;
    for (int i = 0; i < K; i++) mx = fmaxf(mx, fabsf(src[i]));
    sc[oc] = mx / 7.0f + 1e-12f;
}

__global__ void k_packB(const float* __restrict__ w, const float* __restrict__ sc,
                        int* __restrict__ pk, int total, int CIN) {
    int tid = blockIdx.x * blockDim.x + threadIdx.x;
    if (tid >= total) return;
    int v    = tid & 7;
    int rest = tid >> 3;
    int lane = rest & 31; rest >>= 5;
    int t    = rest % 9;
    int ocg  = rest / 9;
    int oc   = ocg * 16 + (lane & 15);
    int hi   = lane >> 4;
    int kb   = ((v >> 2) * 32) + ((v & 3) * 4) + hi * 16;
    int ky = t / 3, kx = t % 3;
    float s = sc[oc];
    uint32_t d = 0;
    for (int j = 0; j < 4; j++) {
        int ci = kb + j;
        float wv = w[(((size_t)oc * CIN + ci) * 3 + ky) * 3 + kx];
        int q = (int)rintf(wv / s);
        q = q < -7 ? -7 : (q > 7 ? 7 : q);
        d |= ((uint32_t)(uint8_t)(int8_t)q) << (8 * j);
    }
    pk[tid] = (int)d;
}

__global__ void k_packW4(const float* __restrict__ w, const float* __restrict__ sc,
                         int8_t* __restrict__ pk) {
    int tid = blockIdx.x * blockDim.x + threadIdx.x;
    if (tid >= 864) return;
    int c = tid & 31;
    int rest = tid >> 5;
    int t = rest % 9, oc = rest / 9;
    int ky = t / 3, kx = t % 3;
    float s = sc[oc];
    float wv = w[((oc * 32 + c) * 3 + ky) * 3 + kx];
    int q = (int)rintf(wv / s);
    q = q < -7 ? -7 : (q > 7 ? 7 : q);
    pk[((oc * 3 + ky) * 3 + kx) * 32 + c] = (int8_t)q;
}

__global__ void k_conv1(const float* __restrict__ inp, const float* __restrict__ wd,
                        const float* __restrict__ b1, const float* __restrict__ g,
                        const float* __restrict__ be, const float* __restrict__ mn,
                        const float* __restrict__ vr, uint8_t* __restrict__ out) {
    int gid = blockIdx.x * blockDim.x + threadIdx.x;
    int oc  = gid & 63;
    int pix = gid >> 6;
    int x = pix & 255;
    int y = (pix >> 8) & 255;
    int b = pix >> 16;
    float acc = 0.f;
    const float* wo = wd + oc * 75;
    for (int c = 0; c < 3; c++) {
        const float* ip = inp + ((size_t)(b * 3 + c) * 256) * 256;
        for (int ky = 0; ky < 5; ky++) {
            int iy = y + ky - 2;
            if (iy < 0 || iy > 255) continue;
            for (int kx = 0; kx < 5; kx++) {
                int ix = x + kx - 2;
                if (ix < 0 || ix > 255) continue;
                float xv = fmaxf(ip[iy * 256 + ix], 0.f);
                float a = fminf(fmaxf(rintf(xv * (1.0f / S_IO)), 0.f), 255.f) * S_IO;
                acc += a * wo[(c * 5 + ky) * 5 + kx];
            }
        }
    }
    acc += b1[oc];
    float inv = g[oc] / sqrtf(vr[oc] + BN_EPS);
    acc = acc * inv + (be[oc] - mn[oc] * inv);
    acc = fmaxf(acc, 0.f);
    int q = (int)fminf(fmaxf(rintf(acc * (1.0f / S_ACT)), 0.f), 15.f);
    out[((size_t)(b * 258 + y + 1) * 258 + (x + 1)) * 64 + oc] = (uint8_t)q;
}

__global__ void k_convB(const uint8_t* __restrict__ actIn, const int* __restrict__ wpk,
                        const float* __restrict__ wsc, const float* __restrict__ bias,
                        const float* __restrict__ g, const float* __restrict__ be,
                        const float* __restrict__ mn, const float* __restrict__ vr,
                        uint8_t* __restrict__ actOut, int outC) {
    int lane = threadIdx.x & 31;
    int ocg  = threadIdx.x >> 5;
    int blk  = blockIdx.x;
    int b    = blk >> 10;
    int rem  = blk & 1023;
    int y    = rem >> 2;
    int x0   = (rem & 3) << 6;
    int m    = lane & 15;
    int hi   = lane >> 4;

    v8i acc0 = {0, 0, 0, 0, 0, 0, 0, 0};
    v8i acc1 = acc0, acc2 = acc0, acc3 = acc0;

#pragma unroll
    for (int t = 0; t < 9; t++) {
        int dy = t / 3, dx = t % 3;
        const int* wp = wpk + ((((ocg * 9 + t) * 32) + lane) << 3);
        int4 w0 = *(const int4*)(wp);
        int4 w1 = *(const int4*)(wp + 4);
        v8i bb;
        bb[0] = w0.x; bb[1] = w0.y; bb[2] = w0.z; bb[3] = w0.w;
        bb[4] = w1.x; bb[5] = w1.y; bb[6] = w1.z; bb[7] = w1.w;

        const uint8_t* prow = actIn +
            (((size_t)(b * 258 + y + dy) * 258) + (x0 + m + dx)) * 64 + hi * 16;
#pragma unroll
        for (int u = 0; u < 4; u++) {
            const uint8_t* p = prow + (size_t)u * 16 * 64;
            v8i a;
            uint4 q;
            q = *(const uint4*)(p +  0); a[0] = (int)q.x; a[1] = (int)q.y; a[2] = (int)q.z; a[3] = (int)q.w;
            q = *(const uint4*)(p + 32); a[4] = (int)q.x; a[5] = (int)q.y; a[6] = (int)q.z; a[7] = (int)q.w;
            if (u == 0)
                acc0 = __builtin_amdgcn_wmma_i32_16x16x64_iu8(false, a, true, bb,
                                                              acc0, false, false);
            else if (u == 1)
                acc1 = __builtin_amdgcn_wmma_i32_16x16x64_iu8(false, a, true, bb,
                                                              acc1, false, false);
            else if (u == 2)
                acc2 = __builtin_amdgcn_wmma_i32_16x16x64_iu8(false, a, true, bb,
                                                              acc2, false, false);
            else
                acc3 = __builtin_amdgcn_wmma_i32_16x16x64_iu8(false, a, true, bb,
                                                              acc3, false, false);
        }
    }

    int n  = lane & 15;
    int oc = ocg * 16 + n;
    float sw  = wsc[oc] * S_ACT;
    float inv = g[oc] / sqrtf(vr[oc] + BN_EPS);
    float bet = be[oc] - mn[oc] * inv;
    float bia = bias[oc];
    v8i accs[4] = {acc0, acc1, acc2, acc3};
#pragma unroll
    for (int u = 0; u < 4; u++) {
        size_t rowbase = ((size_t)(b * 258 + y + 1) * 258);
#pragma unroll
        for (int r = 0; r < 8; r++) {
            int pm = r + hi * 8;
            float val = (float)accs[u][r] * sw + bia;
            val = val * inv + bet;
            val = fmaxf(val, 0.f);
            int q4 = (int)fminf(fmaxf(rintf(val * (1.0f / S_ACT)), 0.f), 15.f);
            actOut[(rowbase + (x0 + u * 16 + pm + 1)) * outC + oc] = (uint8_t)q4;
        }
    }
}

__global__ void k_conv4(const uint8_t* __restrict__ act3, const int8_t* __restrict__ wq,
                        const float* __restrict__ wsc, const float* __restrict__ b4,
                        float* __restrict__ out) {
    int tid  = blockIdx.x * blockDim.x + threadIdx.x;
    int ox   = tid % 768;
    int rest = tid / 768;
    int oy   = rest % 768;
    int b    = rest / 768;
    int acc0 = 0, acc1 = 0, acc2 = 0;
    for (int ky = 0; ky < 3; ky++) {
        int iyu = oy + ky - 1;
        int iy  = (iyu + 3) / 3 - 1;
        for (int kx = 0; kx < 3; kx++) {
            int ixu = ox + kx - 1;
            int ix  = (ixu + 3) / 3 - 1;
            const uint8_t* p = act3 +
                (((size_t)(b * 258 + iy + 1) * 258) + (ix + 1)) * 32;
            const int8_t* w0 = wq + ((0 * 3 + ky) * 3 + kx) * 32;
            const int8_t* w1 = wq + ((1 * 3 + ky) * 3 + kx) * 32;
            const int8_t* w2 = wq + ((2 * 3 + ky) * 3 + kx) * 32;
#pragma unroll
            for (int c = 0; c < 32; c++) {
                int a = (int)p[c];
                acc0 += a * (int)w0[c];
                acc1 += a * (int)w1[c];
                acc2 += a * (int)w2[c];
            }
        }
    }
    int accs[3] = {acc0, acc1, acc2};
    for (int oc = 0; oc < 3; oc++) {
        float r = (float)accs[oc] * (S_ACT * wsc[oc]) + b4[oc];
        r = fmaxf(r, 0.f);
        float q = fminf(fmaxf(rintf(r * (1.0f / S_IO)), 0.f), 255.f);
        out[(((size_t)(b * 3 + oc) * 768) + oy) * 768 + ox] = q * S_IO;
    }
}

extern "C" void kernel_launch(void* const* d_in, const int* in_sizes, int n_in,
                              void* d_out, int out_size, void* d_ws, size_t ws_size,
                              hipStream_t stream) {
    const float* inp = (const float*)d_in[0];
    const float* w1  = (const float*)d_in[1];
    const float* b1  = (const float*)d_in[2];
    const float* g1  = (const float*)d_in[3];
    const float* be1 = (const float*)d_in[4];
    const float* m1  = (const float*)d_in[5];
    const float* v1  = (const float*)d_in[6];
    const float* w2  = (const float*)d_in[7];
    const float* b2  = (const float*)d_in[8];
    const float* g2  = (const float*)d_in[9];
    const float* be2 = (const float*)d_in[10];
    const float* m2  = (const float*)d_in[11];
    const float* v2  = (const float*)d_in[12];
    const float* w3  = (const float*)d_in[13];
    const float* b3  = (const float*)d_in[14];
    const float* g3  = (const float*)d_in[15];
    const float* be3 = (const float*)d_in[16];
    const float* m3  = (const float*)d_in[17];
    const float* v3  = (const float*)d_in[18];
    const float* w4  = (const float*)d_in[19];
    const float* b4  = (const float*)d_in[20];

    uint8_t* ws   = (uint8_t*)d_ws;
    uint8_t* act1 = ws + OFF_ACT1;
    uint8_t* act2 = ws + OFF_ACT2;
    uint8_t* act3 = ws + OFF_ACT3;
    float*   w1d  = (float*)(ws + OFF_W1D);
    int*     w2p  = (int*)(ws + OFF_W2P);
    float*   s2   = (float*)(ws + OFF_S2);
    int*     w3p  = (int*)(ws + OFF_W3P);
    float*   s3   = (float*)(ws + OFF_S3);
    int8_t*  w4q  = (int8_t*)(ws + OFF_W4Q);
    float*   s4   = (float*)(ws + OFF_S4);
    float*   out  = (float*)d_out;

    {
        size_t n16 = ZERO_BYTES / 16;
        int blocks = (int)((n16 + 255) / 256);
        k_zero<<<blocks, 256, 0, stream>>>((uint4*)ws, n16);
    }
    k_w1_dequant<<<1, 64, 0, stream>>>(w1, w1d);
    k_wscale<<<1, 64, 0, stream>>>(w2, s2, 64 * 9);
    k_wscale<<<1, 32, 0, stream>>>(w3, s3, 64 * 9);
    k_wscale<<<1, 3, 0, stream>>>(w4, s4, 32 * 9);
    k_packB<<<(4 * 9 * 32 * 8 + 255) / 256, 256, 0, stream>>>(w2, s2, w2p,
                                                              4 * 9 * 32 * 8, 64);
    k_packB<<<(2 * 9 * 32 * 8 + 255) / 256, 256, 0, stream>>>(w3, s3, w3p,
                                                              2 * 9 * 32 * 8, 64);
    k_packW4<<<4, 256, 0, stream>>>(w4, s4, w4q);

    k_conv1<<<65536, 256, 0, stream>>>(inp, w1d, b1, g1, be1, m1, v1, act1);

    k_convB<<<4096, 128, 0, stream>>>(act1, w2p, s2, b2, g2, be2, m2, v2, act2, 64);

    k_convB<<<4096, 64, 0, stream>>>(act2, w3p, s3, b3, g3, be3, m3, v3, act3, 32);

    k_conv4<<<(4 * 768 * 768) / 256, 256, 0, stream>>>(act3, w4q, s4, b4, out);
}
